// SyncAwareAttention_76218489635255
// MI455X (gfx1250) — compile-verified, matrix operand order checked by reading, not run on hardware
//
#include <hip/hip_runtime.h>

// ---------------------------------------------------------------------------
// SyncAwareAttention for MI455X (gfx1250), bf16 WMMA pipeline.
// B=2, S=2048, E=1024, H=16, DH=64, HID=256.
// Attention computes transposed score tiles (scoresT = K x Q^T) so that the
// WMMA C layout puts query-row i in the lane index: softmax reductions become
// in-lane + one shfl_xor(16), and the p-tile lands directly in the WMMA
// A-operand layout for the ctx GEMM (no LDS bounce, no barriers).
// GEMM A-tile staging uses GLOBAL_LOAD_ASYNC_TO_LDS_B128 (ASYNCcnt) when the
// toolchain exposes the builtin.
// ---------------------------------------------------------------------------

#define B_  2
#define S_  2048
#define E_  1024
#define H_  16
#define DH_ 64
#define HID_ 256

#if defined(__gfx1250__) && __has_builtin(__builtin_amdgcn_global_load_async_to_lds_b128)
#define USE_ASYNC_LDS 1
#else
#define USE_ASYNC_LDS 0
#endif

typedef __bf16 v8bf  __attribute__((ext_vector_type(8)));
typedef __bf16 v16bf __attribute__((ext_vector_type(16)));
typedef float  v8f   __attribute__((ext_vector_type(8)));
typedef int    v4i   __attribute__((ext_vector_type(4)));

#if USE_ASYNC_LDS
typedef __attribute__((address_space(1))) v4i* gp4;  // global int4*
typedef __attribute__((address_space(3))) v4i* lp4;  // LDS int4*
#endif

// Build a 16x32 (16-bit) WMMA A/B fragment from a row of 32 contiguous bf16
// K-values, following the CDNA5 layout: lanes 0-15 hold K={0..7,16..23},
// lanes 16-31 hold K={8..15,24..31}.  g = lane>>4.
__device__ inline v16bf frag_from_row(const __bf16* row, int g) {
  v8bf lo = *(const v8bf*)(row + g * 8);
  v8bf hi = *(const v8bf*)(row + 16 + g * 8);
  v16bf r;
#pragma unroll
  for (int i = 0; i < 8; ++i) { r[i] = lo[i]; r[i + 8] = hi[i]; }
  return r;
}

__device__ inline v8f wmma_bf16(v16bf a, v16bf b, v8f c) {
  return __builtin_amdgcn_wmma_f32_16x16x32_bf16(false, a, false, b, (short)0, c,
                                                 false, false);
}

// ---------------------------------------------------------------------------
// Zero-fill (float4 granularity)
// ---------------------------------------------------------------------------
__global__ void zero_f4(float4* p, int n4) {
  int i = blockIdx.x * blockDim.x + threadIdx.x;
  int stride = gridDim.x * blockDim.x;
  float4 z = make_float4(0.f, 0.f, 0.f, 0.f);
  for (; i < n4; i += stride) p[i] = z;
}

// ---------------------------------------------------------------------------
// f32 -> bf16 conversion
// ---------------------------------------------------------------------------
__global__ void f32_to_bf16(const float* __restrict__ src, __bf16* __restrict__ dst,
                            int n) {
  int i = blockIdx.x * blockDim.x + threadIdx.x;
  int stride = gridDim.x * blockDim.x;
  for (; i < n; i += stride) dst[i] = (__bf16)src[i];
}

// ---------------------------------------------------------------------------
// Gate MLP: sw[b,h] = sigmoid(relu(sync @ Ws1 + bs1) @ Ws2 + bs2)
// ---------------------------------------------------------------------------
__global__ void gate_kernel(const float* __restrict__ sync,
                            const float* __restrict__ Ws1,
                            const float* __restrict__ bs1,
                            const float* __restrict__ Ws2,
                            const float* __restrict__ bs2,
                            float* __restrict__ swOut) {
  __shared__ float hid[HID_];
  int b = blockIdx.x;
  int j = threadIdx.x;
  float acc = bs1[j];
  for (int e = 0; e < E_; ++e) acc += sync[b * E_ + e] * Ws1[e * HID_ + j];
  hid[j] = fmaxf(acc, 0.f);
  __syncthreads();
  if (j < H_) {
    float a = bs2[j];
    for (int t = 0; t < HID_; ++t) a += hid[t] * Ws2[t * H_ + j];
    swOut[b * H_ + j] = 1.f / (1.f + __expf(-a));
  }
}

// ---------------------------------------------------------------------------
// Generic bf16 GEMM:  C[M,N] = A[M,K] @ W[K,N] + bias
// Block tile 128x64, K-step 32, 256 threads = 8 waves, wave w owns 16 rows.
// MODE 0: bf16 row-major out
// MODE 1: bf16 out transposed per head: vT[b][h][d][s]  (for V)
// MODE 2: f32 row-major out (final projection into d_out)
// ---------------------------------------------------------------------------
template <int MODE>
__global__ void gemm_bf16(const __bf16* __restrict__ A,
                          const __bf16* __restrict__ W,
                          const float* __restrict__ bias, void* __restrict__ out,
                          int M, int N, int K) {
  __shared__ __bf16 As[128][32];
  __shared__ __bf16 Wt[64][32];  // transposed: Wt[n][k]

  const int nBase = blockIdx.x * 64;
  const int mBase = blockIdx.y * 128;
  const int w = threadIdx.x >> 5;
  const int lane = threadIdx.x & 31;
  const int g = lane >> 4;
  const int mn = lane & 15;

  v8f acc[4];
#pragma unroll
  for (int nt = 0; nt < 4; ++nt)
#pragma unroll
    for (int r = 0; r < 8; ++r) acc[nt][r] = 0.f;

  const int kW = threadIdx.x >> 3;          // 0..31
  const int n0 = (threadIdx.x & 7) * 8;     // 0,8,..,56
  const int rA = threadIdx.x >> 1;          // 0..127
  const int cA = (threadIdx.x & 1) * 16;

  for (int kb = 0; kb < K / 32; ++kb) {
    const __bf16* wsrc = W + (size_t)(kb * 32 + kW) * N + nBase + n0;
    const __bf16* asrc = A + (size_t)(mBase + rA) * K + kb * 32 + cA;
    // prefetch next K-tile (global_prefetch_b8)
    if (kb + 1 < K / 32) {
      __builtin_prefetch(wsrc + (size_t)32 * N, 0, 1);
      __builtin_prefetch(asrc + 32, 0, 1);
    }
    // A tile (128 x 32): direct global -> LDS via async DMA when available
#if USE_ASYNC_LDS
    {
      __builtin_amdgcn_global_load_async_to_lds_b128((gp4)asrc, (lp4)&As[rA][cA],
                                                     0, 0);
      __builtin_amdgcn_global_load_async_to_lds_b128((gp4)(asrc + 8),
                                                     (lp4)&As[rA][cA + 8], 0, 0);
    }
#else
    {
      *(v8bf*)(&As[rA][cA])     = *(const v8bf*)(asrc);
      *(v8bf*)(&As[rA][cA + 8]) = *(const v8bf*)(asrc + 8);
    }
#endif
    // load W tile (32 x 64) and transpose into LDS (through VGPRs)
    {
      v8bf wv = *(const v8bf*)wsrc;
#pragma unroll
      for (int i = 0; i < 8; ++i) Wt[n0 + i][kW] = wv[i];
    }
#if USE_ASYNC_LDS
#if __has_builtin(__builtin_amdgcn_s_wait_asynccnt)
    __builtin_amdgcn_s_wait_asynccnt(0);
#else
    asm volatile("s_wait_asynccnt 0x0" ::: "memory");
#endif
#endif
    __syncthreads();

    v16bf af = frag_from_row(&As[w * 16 + mn][0], g);
#pragma unroll
    for (int nt = 0; nt < 4; ++nt) {
      v16bf bf_ = frag_from_row(&Wt[nt * 16 + mn][0], g);
      acc[nt] = wmma_bf16(af, bf_, acc[nt]);
    }
    __syncthreads();
  }

  // epilogue: C layout -> lane holds (row = g*8+r, col = nt*16 + mn)
#pragma unroll
  for (int nt = 0; nt < 4; ++nt) {
    int col = nBase + nt * 16 + mn;
    float bv = bias[col];
#pragma unroll
    for (int r = 0; r < 8; ++r) {
      int row = mBase + w * 16 + g * 8 + r;
      float val = acc[nt][r] + bv;
      if (MODE == 2) {
        ((float*)out)[(size_t)row * N + col] = val;
      } else if (MODE == 0) {
        ((__bf16*)out)[(size_t)row * N + col] = (__bf16)val;
      } else {  // MODE 1: vT[b][h][d][s]
        int hh = col >> 6, d = col & 63;
        int bb = row >> 11, s = row & (S_ - 1);
        ((__bf16*)out)[(((size_t)bb * H_ + hh) * DH_ + d) * S_ + s] = (__bf16)val;
      }
    }
  }
}

// ---------------------------------------------------------------------------
// scoresT tile: 16 key rows (jt*16 + mn as WMMA-A) x 16 query rows (WMMA-B).
// C layout: lane L holds i = L&15, j-values r + 8*(L>>4).
// ---------------------------------------------------------------------------
__device__ inline v8f score_t_tile(const __bf16* __restrict__ k, v16bf qb0,
                                   v16bf qb1, int b, int h, int jrow, int g) {
  const __bf16* krow = k + (size_t)(b * S_ + jrow) * E_ + h * DH_;
  v16bf ka0 = frag_from_row(krow, g);
  v16bf ka1 = frag_from_row(krow + 32, g);
  v8f sc;
#pragma unroll
  for (int r = 0; r < 8; ++r) sc[r] = 0.f;
  sc = wmma_bf16(ka0, qb0, sc);
  sc = wmma_bf16(ka1, qb1, sc);
  return sc;
}

// ---------------------------------------------------------------------------
// Attention pass 1: streaming row max / sum-exp.  One wave32 per block.
// ---------------------------------------------------------------------------
__global__ void attn_pass1(const __bf16* __restrict__ q,
                           const __bf16* __restrict__ k,
                           const float* __restrict__ sw,
                           float* __restrict__ mOut, float* __restrict__ lOut) {
  const int it = blockIdx.x, h = blockIdx.y, b = blockIdx.z;
  const int lane = threadIdx.x;
  const int g = lane >> 4;
  const int mn = lane & 15;

  const __bf16* qrow = q + (size_t)(b * S_ + it * 16 + mn) * E_ + h * DH_;
  v16bf qb0 = frag_from_row(qrow, g);
  v16bf qb1 = frag_from_row(qrow + 32, g);
  const float scaleGate = sw[b * H_ + h] * 0.125f;  // gate / sqrt(64)

  float mrun = -3.0e38f, lrun = 0.f;

  for (int jt = 0; jt < S_ / 16; ++jt) {
    v8f sc = score_t_tile(k, qb0, qb1, b, h, jt * 16 + mn, g);
    float t[8];
    float tmax = -3.0e38f;
#pragma unroll
    for (int r = 0; r < 8; ++r) {
      t[r] = sc[r] * scaleGate;
      tmax = fmaxf(tmax, t[r]);
    }
    tmax = fmaxf(tmax, __shfl_xor(tmax, 16));
    float mnew = fmaxf(mrun, tmax);
    float esum = 0.f;
#pragma unroll
    for (int r = 0; r < 8; ++r) esum += __expf(t[r] - mnew);
    esum += __shfl_xor(esum, 16);
    lrun = lrun * __expf(mrun - mnew) + esum;
    mrun = mnew;
  }
  if (g == 0) {
    size_t idx = (size_t)(b * H_ + h) * S_ + it * 16 + mn;
    mOut[idx] = mrun;
    lOut[idx] = lrun;
  }
}

// ---------------------------------------------------------------------------
// Attention pass 2: p = exp(s - m)/l; atomic-accumulate p/H into attn-mean;
// ctx += p @ v via WMMA.  p tile is already in the WMMA A-operand layout.
// ---------------------------------------------------------------------------
__global__ void attn_pass2(const __bf16* __restrict__ q,
                           const __bf16* __restrict__ k,
                           const __bf16* __restrict__ vT,
                           const float* __restrict__ sw,
                           const float* __restrict__ mIn,
                           const float* __restrict__ lIn,
                           float* __restrict__ meanOut,
                           __bf16* __restrict__ ctx) {
  const int it = blockIdx.x, h = blockIdx.y, b = blockIdx.z;
  const int lane = threadIdx.x;
  const int g = lane >> 4;
  const int mn = lane & 15;

  const __bf16* qrow = q + (size_t)(b * S_ + it * 16 + mn) * E_ + h * DH_;
  v16bf qb0 = frag_from_row(qrow, g);
  v16bf qb1 = frag_from_row(qrow + 32, g);
  const float scaleGate = sw[b * H_ + h] * 0.125f;
  const float invH = 1.0f / (float)H_;

  const size_t mlIdx = (size_t)(b * H_ + h) * S_ + it * 16 + mn;
  const float mrow = mIn[mlIdx];
  const float linv = 1.0f / lIn[mlIdx];
  const int i_glob = it * 16 + mn;
  float* meanRow = meanOut + ((size_t)b * S_ + i_glob) * S_;

  v8f cacc[4];
#pragma unroll
  for (int nt = 0; nt < 4; ++nt)
#pragma unroll
    for (int r = 0; r < 8; ++r) cacc[nt][r] = 0.f;

  for (int jc = 0; jc < S_ / 32; ++jc) {
    v16bf pa;
#pragma unroll
    for (int half = 0; half < 2; ++half) {
      int jb = jc * 32 + half * 16;
      v8f sc = score_t_tile(k, qb0, qb1, b, h, jb + mn, g);
#pragma unroll
      for (int r = 0; r < 8; ++r) {
        float p = __expf(sc[r] * scaleGate - mrow) * linv;
        int j = jb + g * 8 + r;
        atomicAdd(&meanRow[j], p * invH);
        pa[half * 8 + r] = (__bf16)p;  // elements 0..7 / 8..15 = A-frag halves
      }
    }
#pragma unroll
    for (int nt = 0; nt < 4; ++nt) {
      const __bf16* vrow =
          vT + (size_t)((b * H_ + h) * DH_ + nt * 16 + mn) * S_ + jc * 32;
      v16bf vb = frag_from_row(vrow, g);
      cacc[nt] = wmma_bf16(pa, vb, cacc[nt]);
    }
  }

#pragma unroll
  for (int nt = 0; nt < 4; ++nt)
#pragma unroll
    for (int r = 0; r < 8; ++r) {
      size_t row = (size_t)b * S_ + it * 16 + g * 8 + r;
      int col = h * DH_ + nt * 16 + mn;
      ctx[row * E_ + col] = (__bf16)cacc[nt][r];
    }
}

// ---------------------------------------------------------------------------
// Host-side launcher
// ---------------------------------------------------------------------------
extern "C" void kernel_launch(void* const* d_in, const int* in_sizes, int n_in,
                              void* d_out, int out_size, void* d_ws, size_t ws_size,
                              hipStream_t stream) {
  const float* x    = (const float*)d_in[0];
  const float* sync = (const float*)d_in[1];
  const float* Wq   = (const float*)d_in[2];
  const float* bq   = (const float*)d_in[3];
  const float* Wk   = (const float*)d_in[4];
  const float* bk   = (const float*)d_in[5];
  const float* Wv   = (const float*)d_in[6];
  const float* bv   = (const float*)d_in[7];
  const float* Ws1  = (const float*)d_in[8];
  const float* bs1  = (const float*)d_in[9];
  const float* Ws2  = (const float*)d_in[10];
  const float* bs2  = (const float*)d_in[11];
  const float* Wo   = (const float*)d_in[12];
  const float* bo   = (const float*)d_in[13];

  float* out_proj = (float*)d_out;                                 // [B,S,E]
  float* out_mean = (float*)d_out + (size_t)B_ * S_ * E_;          // [B,S,S]

  // workspace layout (bytes)
  char* ws = (char*)d_ws;
  size_t off = 0;
  auto alloc = [&](size_t bytes) { size_t o = off; off += (bytes + 255) & ~(size_t)255; return o; };
  const size_t nX = (size_t)B_ * S_ * E_;      // 4,194,304
  const size_t nW = (size_t)E_ * E_;           // 1,048,576

  size_t off_sw  = alloc(B_ * H_ * sizeof(float));
  size_t off_xbf = alloc(nX * 2);
  size_t off_wq  = alloc(nW * 2);
  size_t off_wk  = alloc(nW * 2);
  size_t off_wv  = alloc(nW * 2);
  size_t off_wo  = alloc(nW * 2);
  size_t off_q   = alloc(nX * 2);
  size_t off_k   = alloc(nX * 2);
  size_t off_vT  = alloc(nX * 2);
  size_t off_m   = alloc((size_t)B_ * H_ * S_ * sizeof(float));
  size_t off_l   = alloc((size_t)B_ * H_ * S_ * sizeof(float));
  size_t off_ctx = alloc(nX * 2);
  (void)ws_size;

  float*  sw_g   = (float*)(ws + off_sw);
  __bf16* xbf    = (__bf16*)(ws + off_xbf);
  __bf16* wq_bf  = (__bf16*)(ws + off_wq);
  __bf16* wk_bf  = (__bf16*)(ws + off_wk);
  __bf16* wv_bf  = (__bf16*)(ws + off_wv);
  __bf16* wo_bf  = (__bf16*)(ws + off_wo);
  __bf16* q_bf   = (__bf16*)(ws + off_q);
  __bf16* k_bf   = (__bf16*)(ws + off_k);
  __bf16* vT_bf  = (__bf16*)(ws + off_vT);
  float*  m_buf  = (float*)(ws + off_m);
  float*  l_buf  = (float*)(ws + off_l);
  __bf16* ctx_bf = (__bf16*)(ws + off_ctx);

  // 0) zero the attn-mean region (poisoned / accumulated via atomics)
  zero_f4<<<2048, 256, 0, stream>>>((float4*)out_mean, (int)((size_t)B_ * S_ * S_ / 4));

  // 1) gate MLP
  gate_kernel<<<B_, HID_, 0, stream>>>(sync, Ws1, bs1, Ws2, bs2, sw_g);

  // 2) f32 -> bf16 conversions
  f32_to_bf16<<<2048, 256, 0, stream>>>(x, xbf, (int)nX);
  f32_to_bf16<<<1024, 256, 0, stream>>>(Wq, wq_bf, (int)nW);
  f32_to_bf16<<<1024, 256, 0, stream>>>(Wk, wk_bf, (int)nW);
  f32_to_bf16<<<1024, 256, 0, stream>>>(Wv, wv_bf, (int)nW);
  f32_to_bf16<<<1024, 256, 0, stream>>>(Wo, wo_bf, (int)nW);

  // 3) QKV projections (M=4096, N=K=1024)
  dim3 ggrid(E_ / 64, (B_ * S_) / 128);
  gemm_bf16<0><<<ggrid, 256, 0, stream>>>(xbf, wq_bf, bq, q_bf, B_ * S_, E_, E_);
  gemm_bf16<0><<<ggrid, 256, 0, stream>>>(xbf, wk_bf, bk, k_bf, B_ * S_, E_, E_);
  gemm_bf16<1><<<ggrid, 256, 0, stream>>>(xbf, wv_bf, bv, vT_bf, B_ * S_, E_, E_);

  // 4) attention: two-pass online softmax + ctx
  dim3 agrid(S_ / 16, H_, B_);
  attn_pass1<<<agrid, 32, 0, stream>>>(q_bf, k_bf, sw_g, m_buf, l_buf);
  attn_pass2<<<agrid, 32, 0, stream>>>(q_bf, k_bf, vT_bf, sw_g, m_buf, l_buf,
                                       out_mean, ctx_bf);

  // 5) output projection -> f32 d_out
  gemm_bf16<2><<<ggrid, 256, 0, stream>>>(ctx_bf, wo_bf, bo, out_proj, B_ * S_, E_, E_);
}
